// RPN_43946105372763
// MI455X (gfx1250) — matrix operand order repaired and checked by reading, not compiled, not run on hardware
//
#include <hip/hip_runtime.h>

// RPN head for MI455X (gfx1250): implicit-GEMM 3x3 conv via v_wmma_f32_16x16x32_f16
// with TDM (tensor_load_to_lds) double-buffered weight staging, fused ReLU +
// 1x1 heads (second chained WMMA) + anchor decode + sigmoid, then radix-select
// top-1000, bitonic sort, greedy NMS.

typedef __attribute__((ext_vector_type(16))) _Float16 v16h;
typedef __attribute__((ext_vector_type(8)))  float    v8f;
typedef __attribute__((ext_vector_type(4)))  unsigned v4u;
typedef __attribute__((ext_vector_type(8)))  int      v8i;
typedef __attribute__((ext_vector_type(4)))  int      v4i;

#define A_TOTAL   261888
#define PRE_NMS   1000
#define POST_NMS  300
#define CAND_CAP  4096

#if defined(__gfx1250__) && __has_builtin(__builtin_amdgcn_tensor_load_to_lds)
#define RPN_USE_TDM 1
#else
#define RPN_USE_TDM 0
#endif

union Frag16 { v16h h; unsigned u[8]; };

__device__ __constant__ int   g_H[5]      = {256, 128, 64, 32, 16};
__device__ __constant__ int   g_stride[5] = {4, 8, 16, 32, 64};
__device__ __constant__ float g_size[5]   = {32.f, 64.f, 128.f, 256.f, 512.f};
__device__ __constant__ int   g_abase[5]  = {0, 196608, 245760, 258048, 261120};

// ---------------------------------------------------------------------------
// TDM helpers: DMA one contiguous 16 KB weight chunk (8192 f16) into LDS.
// D# packed per CDNA5 ISA 8.3/8.4: count=1, data_size=2B, tile = 8192 x 1.
// ---------------------------------------------------------------------------
__device__ __forceinline__ unsigned lds_offset_of(const void* p) {
    return (unsigned)(unsigned long long)(__attribute__((address_space(3))) const void*)p;
}

__device__ __forceinline__ void stage_weights_issue(const _Float16* __restrict__ src,
                                                    _Float16* ldsDst, int wave, int tid) {
#if RPN_USE_TDM
    if (wave == 0) {
        unsigned long long ga = (unsigned long long)(const void*)src;
        unsigned lds = lds_offset_of(ldsDst);
        v4u g0;
        g0[0] = 1u;                                    // count=1, user descriptor
        g0[1] = lds;                                   // lds_addr
        g0[2] = (unsigned)(ga & 0xffffffffu);          // global_addr[31:0]
        g0[3] = (unsigned)((ga >> 32) & 0x01ffffffu)   // global_addr[56:32]
              | (2u << 30);                            // type = 2 ("image")
        v8i g1;
        g1[0] = (int)(1u << 16);                       // data_size = 1 -> 2 bytes
        g1[1] = (int)((8192u & 0xffffu) << 16);        // tensor_dim0[15:0]
        g1[2] = (int)(((8192u >> 16) & 0xffffu)        // tensor_dim0[31:16]
              |       (1u << 16));                     // tensor_dim1[15:0] = 1
        g1[3] = (int)(8192u << 16);                    // tile_dim0 = 8192
        g1[4] = 1;                                     // tile_dim1 = 1 (tile_dim2 = 0)
        g1[5] = 8192;                                  // tensor_dim0_stride[31:0]
        g1[6] = (int)((8192u & 0xffffu) << 16);        // dim0_stride hi=0 | dim1_stride[15:0]
        g1[7] = 0;                                     // dim1_stride[47:16]
        v4i gz4 = {0, 0, 0, 0};
        v8i gz8 = {0, 0, 0, 0, 0, 0, 0, 0};
        __builtin_amdgcn_tensor_load_to_lds(g0, g1, gz4, gz4, gz8, 0);
    }
#else
    const unsigned* s = (const unsigned*)src;
    unsigned* d = (unsigned*)ldsDst;
    for (int i = tid; i < 4096; i += 512) d[i] = s[i];
#endif
    (void)wave; (void)tid;
}

__device__ __forceinline__ void stage_weights_wait(int wave) {
#if RPN_USE_TDM
    if (wave == 0) {
#if __has_builtin(__builtin_amdgcn_s_wait_tensorcnt)
        __builtin_amdgcn_s_wait_tensorcnt(0);
#else
        asm volatile("s_wait_tensorcnt 0" ::: "memory");
#endif
    }
#endif
    (void)wave;
}

// ---------------------------------------------------------------------------
// Prep: transpose conv weights into K-major f16 chunks.
// K ordering: k = (ky*3+kx)*256 + ci  (f16 K-pairs ci,ci+1 -> LDS-contiguous)
// wT layout: [kc 0..71][co 0..255][kl 0..31]
// ---------------------------------------------------------------------------
__global__ void rpn_prep_wT(const float* __restrict__ conv_w, _Float16* __restrict__ wT) {
    int i = blockIdx.x * blockDim.x + threadIdx.x;
    if (i >= 72 * 256 * 32) return;
    int kc = i >> 13;
    int r  = i & 8191;
    int co = r >> 5;
    int kl = r & 31;
    int k  = (kc << 5) + kl;
    int kq = k >> 8;
    int ci = k & 255;
    int ky = kq / 3;
    int kx = kq - ky * 3;
    wT[i] = (_Float16)conv_w[co * 2304 + ci * 9 + ky * 3 + kx];
}

// Head weights packed: col n<3 = cls channel n, 3..14 = box channel n-3, 15 = 0.
__global__ void rpn_prep_w2(const float* __restrict__ cls_w, const float* __restrict__ box_w,
                            _Float16* __restrict__ w2) {
    int i = blockIdx.x * blockDim.x + threadIdx.x;
    if (i >= 16 * 256) return;
    int n  = i >> 8;
    int ci = i & 255;
    float v = 0.f;
    if (n < 3)       v = cls_w[n * 256 + ci];
    else if (n < 15) v = box_w[(n - 3) * 256 + ci];
    w2[i] = (_Float16)v;
}

// ---------------------------------------------------------------------------
// Fused conv3x3 + ReLU + (cls|box) heads + decode + sigmoid.
// 512 threads = 16 wave32; one 16-position tile per workgroup.
// Wave w computes output channels [16w,16w+16) with 72 chained WMMAs;
// weight chunks are TDM double-buffered.
// LDS arena (60,416 B):
//   loop phase:  [0,27648) patch | [27648,44032) wbuf0 | [44032,60416) wbuf1
//   tail phase:  [0,8448) xbuf   | [8448,16640) w2buf  | [16640,17664) d2buf
// ---------------------------------------------------------------------------
__global__ __launch_bounds__(512) void rpn_conv_fused(
    const float* __restrict__ p2, const float* __restrict__ p3,
    const float* __restrict__ p4, const float* __restrict__ p5,
    const float* __restrict__ p6,
    const _Float16* __restrict__ wT,      // [72][256][32]
    const _Float16* __restrict__ w2,      // [16][256]
    const float* __restrict__ conv_b,
    const float* __restrict__ cls_b,
    const float* __restrict__ box_b,
    float* __restrict__ scoresWS,         // [2][A_TOTAL]
    float* __restrict__ boxesWS)          // [2][A_TOTAL][4]
{
    __shared__ __align__(16) char smem[60416];
    _Float16* patch = (_Float16*)smem;               // [ky][cx][ci]
    _Float16* wbuf0 = (_Float16*)(smem + 27648);     // [co][kl]
    _Float16* wbuf1 = (_Float16*)(smem + 44032);

    const int tid  = threadIdx.x;
    const int lane = tid & 31;
    const int wave = tid >> 5;
    const int g    = lane >> 4;
    const int mrow = lane & 15;
    const int ncol = lane & 15;

    // --- map blockIdx -> (batch, level, y, x0) ---
    int idx = blockIdx.x;
    int b   = idx / 5456;
    int t   = idx - b * 5456;
    int lev, tlbase;
    if      (t >= 5440) { lev = 4; tlbase = 5440; }
    else if (t >= 5376) { lev = 3; tlbase = 5376; }
    else if (t >= 5120) { lev = 2; tlbase = 5120; }
    else if (t >= 4096) { lev = 1; tlbase = 4096; }
    else                { lev = 0; tlbase = 0;    }
    int tl  = t - tlbase;
    int H   = g_H[lev];
    int tpr = H >> 4;
    int y   = tl / tpr;
    int x0  = (tl - y * tpr) << 4;
    const float* fptr = (lev == 0) ? p2 : (lev == 1) ? p3 : (lev == 2) ? p4 : (lev == 3) ? p5 : p6;

    // --- kick off TDM for weight chunk 0 ---
    stage_weights_issue(wT, wbuf0, wave, tid);

    // --- stage input patch (3 rows x 18 cols x 256 ch), zero-padded borders ---
    for (int i = tid; i < 3 * 18 * 256; i += 512) {
        int ci = i / 54;
        int r  = i - ci * 54;
        int ky = r / 18;
        int cx = r - ky * 18;
        int row = y + ky - 1;
        int col = x0 + cx - 1;
        float v = 0.f;
        if (row >= 0 && row < H && col >= 0 && col < H)
            v = fptr[((size_t)(b * 256 + ci) * H + row) * H + col];
        patch[(ky * 18 + cx) * 256 + ci] = (_Float16)v;
    }
    stage_weights_wait(wave);
    __syncthreads();

    // --- main GEMM: 72 K-chunks of 32, TDM double-buffered ---
    v8f acc = {};
    for (int kc = 0; kc < 72; ++kc) {
        _Float16* cur = (kc & 1) ? wbuf1 : wbuf0;
        _Float16* nxt = (kc & 1) ? wbuf0 : wbuf1;
        if (kc + 1 < 72)
            stage_weights_issue(wT + (size_t)(kc + 1) * 8192, nxt, wave, tid);

        int kq = kc >> 3;            // filter tap 0..8
        int ky = kq / 3;
        int kx = kq - ky * 3;
        int cibase = (kc & 7) << 5;

        Frag16 A, Bm;
        const int arow = (ky * 18 + mrow + kx) * 256 + cibase + (g << 3);
#pragma unroll
        for (int v = 0; v < 8; ++v) {
            int kl = ((v & 3) << 1) + ((v >> 2) << 4);
            A.u[v] = *(const unsigned*)&patch[arow + kl];
        }
        const int brow = ((wave << 4) + ncol) * 32 + (g << 3);
#pragma unroll
        for (int v = 0; v < 8; ++v)
            Bm.u[v] = *(const unsigned*)&cur[brow + ((v & 3) << 1) + ((v >> 2) << 4)];

        acc = __builtin_amdgcn_wmma_f32_16x16x32_f16(
            false, A.h, false, Bm.h, (short)0, acc, false, false);

        if (kc + 1 < 72)
            stage_weights_wait(wave);
        __syncthreads();
    }

    // --- tail phase: alias dead patch region ---
    _Float16* xbuf  = (_Float16*)smem;               // [m][ci] pitch 264
    _Float16* w2buf = (_Float16*)(smem + 8448);      // [n][ci]
    float*    d2buf = (float*)(smem + 16640);        // [m][n]

    {   // bias + ReLU, store intermediate X (f16)
        float bias = conv_b[(wave << 4) + ncol];
#pragma unroll
        for (int r = 0; r < 8; ++r) {
            float v = fmaxf(acc[r] + bias, 0.f);
            int m = r + (g << 3);
            xbuf[m * 264 + (wave << 4) + ncol] = (_Float16)v;
        }
    }
    {   // stage head weights
        const unsigned* src = (const unsigned*)w2;
        unsigned* dst = (unsigned*)w2buf;
        for (int i = tid; i < 16 * 256 / 2; i += 512) dst[i] = src[i];
    }
    __syncthreads();

    // --- head GEMM (16 pos x 256) x (256 x 16) on wave 0 ---
    if (wave == 0) {
        v8f acc2 = {};
#pragma unroll
        for (int kc = 0; kc < 8; ++kc) {
            Frag16 A, Bm;
            int abase = (kc << 5) + (g << 3);
#pragma unroll
            for (int v = 0; v < 8; ++v) {
                int kl = ((v & 3) << 1) + ((v >> 2) << 4);
                A.u[v] = *(const unsigned*)&xbuf[mrow * 264 + abase + kl];
            }
            int bbase = ncol * 256 + (kc << 5) + (g << 3);
#pragma unroll
            for (int v = 0; v < 8; ++v)
                Bm.u[v] = *(const unsigned*)&w2buf[bbase + ((v & 3) << 1) + ((v >> 2) << 4)];
            acc2 = __builtin_amdgcn_wmma_f32_16x16x32_f16(
                false, A.h, false, Bm.h, (short)0, acc2, false, false);
        }
#pragma unroll
        for (int r = 0; r < 8; ++r)
            d2buf[(r + (g << 3)) * 16 + ncol] = acc2[r];
    }
    __syncthreads();

    // --- decode 16 positions x 3 anchors ---
    if (tid < 48) {
        int m = tid / 3;
        int a = tid - m * 3;
        const float sq[3] = {0.70710678f, 1.0f, 1.41421356f};
        float obj = d2buf[m * 16 + a] + cls_b[a];
        float dx = d2buf[m * 16 + 3 + a * 4 + 0] + box_b[a * 4 + 0];
        float dy = d2buf[m * 16 + 3 + a * 4 + 1] + box_b[a * 4 + 1];
        float dw = d2buf[m * 16 + 3 + a * 4 + 2] + box_b[a * 4 + 2];
        float dh = d2buf[m * 16 + 3 + a * 4 + 3] + box_b[a * 4 + 3];
        float s  = (float)g_stride[lev];
        float sz = g_size[lev];
        float ah = sz * sq[a];
        float aw = sz / sq[a];
        float ax = (float)(x0 + m) * s;
        float ay = (float)y * s;
        float cx = dx * aw + ax;
        float cy = dy * ah + ay;
        float pw = expf(dw) * aw;
        float ph = expf(dh) * ah;
        float x1 = fminf(fmaxf(cx - 0.5f * pw, 0.f), 1024.f);
        float y1 = fminf(fmaxf(cy - 0.5f * ph, 0.f), 1024.f);
        float x2 = fminf(fmaxf(cx + 0.5f * pw, 0.f), 1024.f);
        float y2 = fminf(fmaxf(cy + 0.5f * ph, 0.f), 1024.f);
        bool valid = ((x2 - x1) >= 1.0f) && ((y2 - y1) >= 1.0f);
        float score = 1.f / (1.f + expf(-obj));
        float masked = valid ? score : -1e9f;
        int aidx = g_abase[lev] + ((y * H + (x0 + m)) * 3 + a);
        scoresWS[(size_t)b * A_TOTAL + aidx] = masked;
        *(float4*)&boxesWS[((size_t)b * A_TOTAL + aidx) * 4] = make_float4(x1, y1, x2, y2);
    }
}

// ---------------------------------------------------------------------------
// Top-k selection: 16-bit radix histogram -> threshold -> compact -> bitonic.
// ---------------------------------------------------------------------------
__device__ __forceinline__ unsigned sortKey(float f) {
    unsigned u = __float_as_uint(f);
    return (u & 0x80000000u) ? ~u : (u | 0x80000000u);
}

__global__ void rpn_sel_init(unsigned* hist, float* candS, int* candI, int* cnts) {
    int i = blockIdx.x * blockDim.x + threadIdx.x;
    int n = gridDim.x * blockDim.x;
    for (int j = i; j < 2 * 65536; j += n) hist[j] = 0u;
    for (int j = i; j < 2 * CAND_CAP; j += n) { candS[j] = -3.0e38f; candI[j] = 0; }
    if (i < 8) cnts[i] = 0;
}

__global__ void rpn_sel_hist(const float* __restrict__ scores, unsigned* hist) {
    int i = blockIdx.x * blockDim.x + threadIdx.x;
    if (i >= 2 * A_TOTAL) return;
    int b = i / A_TOTAL;
    unsigned bucket = sortKey(scores[i]) >> 16;
    atomicAdd(&hist[b * 65536 + bucket], 1u);
}

__global__ void rpn_sel_thresh(const unsigned* __restrict__ hist, int* cnts) {
    __shared__ unsigned psum[256];
    int b = blockIdx.x;
    int t = threadIdx.x;
    const unsigned* h = hist + b * 65536;
    unsigned s = 0;
    for (int j = 0; j < 256; ++j) s += h[t * 256 + j];
    psum[t] = s;
    __syncthreads();
    if (t == 0) {
        unsigned cum = 0;
        int T = 0;
        for (int seg = 255; seg >= 0; --seg) {
            if (cum + psum[seg] >= (unsigned)PRE_NMS) {
                for (int bk = seg * 256 + 255; bk >= seg * 256; --bk) {
                    cum += h[bk];
                    if (cum >= (unsigned)PRE_NMS) { T = bk; break; }
                }
                break;
            }
            cum += psum[seg];
        }
        cnts[2 + b] = T;
    }
}

__global__ void rpn_sel_compact(const float* __restrict__ scores, int* cnts,
                                float* candS, int* candI) {
    int i = blockIdx.x * blockDim.x + threadIdx.x;
    if (i >= 2 * A_TOTAL) return;
    int b  = i / A_TOTAL;
    int ai = i - b * A_TOTAL;
    float sc = scores[i];
    unsigned bucket = sortKey(sc) >> 16;
    if ((int)bucket >= cnts[2 + b]) {
        int pos = atomicAdd(&cnts[b], 1);
        if (pos < CAND_CAP) {
            candS[b * CAND_CAP + pos] = sc;
            candI[b * CAND_CAP + pos] = ai;
        }
    }
}

__global__ __launch_bounds__(1024) void rpn_sel_sort(
    const float* __restrict__ candS, const int* __restrict__ candI,
    const float* __restrict__ boxesWS,
    float* __restrict__ topS, float* __restrict__ topB)
{
    __shared__ float ss[CAND_CAP];
    __shared__ int   si[CAND_CAP];
    int b = blockIdx.x;
    int t = threadIdx.x;
    for (int i = t; i < CAND_CAP; i += 1024) {
        ss[i] = candS[b * CAND_CAP + i];
        si[i] = candI[b * CAND_CAP + i];
    }
    __syncthreads();
    for (int k = 2; k <= CAND_CAP; k <<= 1) {
        for (int j = k >> 1; j > 0; j >>= 1) {
            for (int i = t; i < CAND_CAP; i += 1024) {
                int p = i ^ j;
                if (p > i) {
                    bool dirBest = ((i & k) == 0);
                    float a0 = ss[i], a1 = ss[p];
                    int   i0 = si[i], i1 = si[p];
                    bool pBefore = (a1 > a0) || (a1 == a0 && i1 < i0);
                    bool doSwap = dirBest ? pBefore : !pBefore;
                    if (doSwap) { ss[i] = a1; ss[p] = a0; si[i] = i1; si[p] = i0; }
                }
            }
            __syncthreads();
        }
    }
    for (int i = t; i < PRE_NMS; i += 1024) {
        float sc = ss[i];
        int   ai = si[i];
        bool sentinel = (sc < -1e30f);
        topS[b * PRE_NMS + i] = sentinel ? -1e9f : sc;
        float4 bx = sentinel ? make_float4(0.f, 0.f, 0.f, 0.f)
                             : *(const float4*)&boxesWS[((size_t)b * A_TOTAL + ai) * 4];
        *(float4*)&topB[(size_t)(b * PRE_NMS + i) * 4] = bx;
    }
}

// ---------------------------------------------------------------------------
// Greedy NMS, one block per image.
// ---------------------------------------------------------------------------
__global__ __launch_bounds__(1024) void rpn_nms(
    const float* __restrict__ topS, const float* __restrict__ topB,
    float* __restrict__ out)
{
    __shared__ float bx1[PRE_NMS], by1[PRE_NMS], bx2[PRE_NMS], by2[PRE_NMS];
    __shared__ float bar[PRE_NMS], bs[PRE_NMS];
    __shared__ int   supp[PRE_NMS];
    __shared__ int   order[POST_NMS];
    __shared__ int   nkept;
    int b = blockIdx.x;
    int t = threadIdx.x;
    for (int i = t; i < PRE_NMS; i += 1024) {
        float4 bx = *(const float4*)&topB[(size_t)(b * PRE_NMS + i) * 4];
        bx1[i] = bx.x; by1[i] = bx.y; bx2[i] = bx.z; by2[i] = bx.w;
        bar[i] = (bx.z - bx.x) * (bx.w - bx.y);
        bs[i]  = topS[b * PRE_NMS + i];
        supp[i] = 0;
    }
    __syncthreads();
    for (int i = 0; i < PRE_NMS; ++i) {
        if (!supp[i]) {
            float X1 = bx1[i], Y1 = by1[i], X2 = bx2[i], Y2 = by2[i], AR = bar[i];
            for (int j = i + 1 + t; j < PRE_NMS; j += 1024) {
                float xx1 = fmaxf(X1, bx1[j]);
                float yy1 = fmaxf(Y1, by1[j]);
                float xx2 = fminf(X2, bx2[j]);
                float yy2 = fminf(Y2, by2[j]);
                float inter = fmaxf(xx2 - xx1, 0.f) * fmaxf(yy2 - yy1, 0.f);
                float iou = inter / (AR + bar[j] - inter + 1e-9f);
                if (iou > 0.7f) supp[j] = 1;
            }
        }
        __syncthreads();
    }
    if (t == 0) {
        int nk = 0;
        for (int i = 0; i < PRE_NMS && nk < POST_NMS; ++i)
            if (!supp[i] && bs[i] > -1e8f) order[nk++] = i;
        nkept = nk;
    }
    __syncthreads();
    for (int r = t; r < POST_NMS; r += 1024) {
        float* o = &out[(size_t)(b * POST_NMS + r) * 5];
        if (r < nkept) {
            int i = order[r];
            o[0] = bx1[i]; o[1] = by1[i]; o[2] = bx2[i]; o[3] = by2[i]; o[4] = bs[i];
        } else {
            o[0] = 0.f; o[1] = 0.f; o[2] = 0.f; o[3] = 0.f; o[4] = 0.f;
        }
    }
}

// ---------------------------------------------------------------------------
extern "C" void kernel_launch(void* const* d_in, const int* in_sizes, int n_in,
                              void* d_out, int out_size, void* d_ws, size_t ws_size,
                              hipStream_t stream) {
    (void)in_sizes; (void)n_in; (void)out_size; (void)ws_size;
    const float* p2     = (const float*)d_in[0];
    const float* p3     = (const float*)d_in[1];
    const float* p4     = (const float*)d_in[2];
    const float* p5     = (const float*)d_in[3];
    const float* p6     = (const float*)d_in[4];
    const float* conv_w = (const float*)d_in[5];
    const float* conv_b = (const float*)d_in[6];
    const float* cls_w  = (const float*)d_in[7];
    const float* cls_b  = (const float*)d_in[8];
    const float* box_w  = (const float*)d_in[9];
    const float* box_b  = (const float*)d_in[10];

    char* ws = (char*)d_ws;
    _Float16* wT     = (_Float16*)(ws + 0);          // 1,179,648 B
    _Float16* w2     = (_Float16*)(ws + 1179648);    //     8,192 B
    float*    scores = (float*)   (ws + 1187840);    // 2,095,104 B
    float*    boxes  = (float*)   (ws + 3282944);    // 8,380,416 B
    unsigned* hist   = (unsigned*)(ws + 11663360);   //   524,288 B
    float*    candS  = (float*)   (ws + 12187648);   //    32,768 B
    int*      candI  = (int*)     (ws + 12220416);   //    32,768 B
    int*      cnts   = (int*)     (ws + 12253184);   //       128 B
    float*    topS   = (float*)   (ws + 12253312);   //     8,000 B
    float*    topB   = (float*)   (ws + 12261312);   //    32,000 B

    rpn_prep_wT<<<(72 * 256 * 32 + 255) / 256, 256, 0, stream>>>(conv_w, wT);
    rpn_prep_w2<<<16, 256, 0, stream>>>(cls_w, box_w, w2);
    rpn_conv_fused<<<2 * 5456, 512, 0, stream>>>(p2, p3, p4, p5, p6, wT, w2,
                                                 conv_b, cls_b, box_b, scores, boxes);
    rpn_sel_init<<<256, 256, 0, stream>>>(hist, candS, candI, cnts);
    rpn_sel_hist<<<(2 * A_TOTAL + 255) / 256, 256, 0, stream>>>(scores, hist);
    rpn_sel_thresh<<<2, 256, 0, stream>>>(hist, cnts);
    rpn_sel_compact<<<(2 * A_TOTAL + 255) / 256, 256, 0, stream>>>(scores, cnts, candS, candI);
    rpn_sel_sort<<<2, 1024, 0, stream>>>(candS, candI, boxes, topS, topB);
    rpn_nms<<<2, 1024, 0, stream>>>(topS, topB, (float*)d_out);
}
